// SequentialGraniteMoeExperts_7756710936690
// MI455X (gfx1250) — matrix operand order repaired and checked by reading, not compiled, not run on hardware
//
#include <hip/hip_runtime.h>
#include <hip/hip_bf16.h>

// ---------------------------------------------------------------------------
// MoE calibration GEMM with fused gather, CDNA5 (gfx1250), split-bf16 WMMA.
//
//   out[j, :] = inputs[batch_index[j], :] @ W[expert_of(j), :, :]^T
//
// f32 emulated as x = hi(x) + lo(x) in bf16, a*b ~= ah*bh + al*bh + ah*bl,
// accumulated f32 by V_WMMA_F32_16X16X32_BF16. The hi/lo split is done ONCE
// per launch into d_ws, so the GEMM hot loop is pure WMMA + b128 loads.
// A-tile staging uses GLOBAL_LOAD_ASYNC_TO_LDS_B128 with double-buffered LDS:
// one barrier per K-chunk, staging overlapped with WMMA compute.
// ---------------------------------------------------------------------------

typedef __bf16  bf16_t;
typedef bf16_t  v16bf __attribute__((ext_vector_type(16)));
typedef bf16_t  v8bf  __attribute__((ext_vector_type(8)));
typedef bf16_t  v4bf  __attribute__((ext_vector_type(4)));
typedef float   v8f   __attribute__((ext_vector_type(8)));

#define DIN   1024                  // K
#define DOUT  2048                  // total N
#define KC    32                    // K per WMMA chunk
#define BT    128                   // threads per block (4 waves)
#define BM    32                    // rows per block (2 M-subtiles/wave)
#define BN    256                   // cols per block (4 N-subtiles/wave)
#define PAD   40                    // LDS row pitch in bf16 (80B = 5x16B)

struct bf2 { bf16_t hi; bf16_t lo; };

__device__ __forceinline__ bf2 split1(float x) {
    bf2 r;
    r.hi = (bf16_t)x;
    r.lo = (bf16_t)(x - (float)r.hi);
    return r;
}

// ---------------- pre-pass: elementwise f32 -> (hi, lo) bf16 ----------------
__global__ __launch_bounds__(256)
void split_f32_bf16(const float* __restrict__ src, bf16_t* __restrict__ hi,
                    bf16_t* __restrict__ lo, long n4) {
    long i = (long)blockIdx.x * 256 + threadIdx.x;
    if (i >= n4) return;
    const float4 v = ((const float4*)src)[i];
    const float f[4] = {v.x, v.y, v.z, v.w};
    v4bf h, l;
#pragma unroll
    for (int j = 0; j < 4; ++j) {
        const bf2 s = split1(f[j]);
        h[j] = s.hi;
        l[j] = s.lo;
    }
    ((v4bf*)hi)[i] = h;
    ((v4bf*)lo)[i] = l;
}

// ---------------- main GEMM on pre-split bf16 data --------------------------
__global__ __launch_bounds__(BT)
void moe_gemm_split(const bf16_t* __restrict__ Xh, const bf16_t* __restrict__ Xl,
                    const bf16_t* __restrict__ Wh, const bf16_t* __restrict__ Wl,
                    const int*   __restrict__ esz, const int* __restrict__ bidx,
                    float*       __restrict__ Y,   int E) {
    __shared__ __align__(16) bf16_t Ah[2][BM][PAD];   // double-buffered
    __shared__ __align__(16) bf16_t Al[2][BM][PAD];
    __shared__ int tok[BM];
    __shared__ int s_expert;

    const int tid   = threadIdx.x;
    const int slot0 = blockIdx.x * BM;      // 32 routed slots
    const int ncol0 = blockIdx.y * BN;      // 256 output columns

    if (tid < BM) tok[tid] = bidx[slot0 + tid];
    if (tid == 0) {
        int acc = 0, e = E - 1;
        for (int i = 0; i < E; ++i) { acc += esz[i]; if (slot0 < acc) { e = i; break; } }
        s_expert = e;
    }
    __syncthreads();

    const int lane = tid & 31;
    const int wave = tid >> 5;
    const int m    = lane & 15;             // row (A) / col (B,C) within 16
    const int h    = lane >> 4;             // half-wave select

    const size_t wexp = (size_t)s_expert * DOUT * DIN;
    const bf16_t* __restrict__ Whb = Wh + wexp;
    const bf16_t* __restrict__ Wlb = Wl + wexp;

    // staging assignment: 4 threads per row, 8 bf16 (16B) per thread
    const int sr = tid >> 2;                // 0..31
    const int sc = (tid & 3) * 8;           // 0,8,16,24
    const size_t srow = (size_t)tok[sr] * DIN + sc;

    // LDS byte offsets (low 32 bits of the generic pointer = LDS address)
    unsigned ldsAh0 = (unsigned)(size_t)&Ah[0][sr][sc];
    unsigned ldsAl0 = (unsigned)(size_t)&Al[0][sr][sc];
    unsigned ldsAh1 = (unsigned)(size_t)&Ah[1][sr][sc];
    unsigned ldsAl1 = (unsigned)(size_t)&Al[1][sr][sc];

#define STAGE_ASYNC(LH, LL, K0)                                               \
    do {                                                                      \
        const bf16_t* gh_ = Xh + srow + (K0);                                 \
        const bf16_t* gl_ = Xl + srow + (K0);                                 \
        asm volatile("global_load_async_to_lds_b128 %0, %2, off\n\t"          \
                     "global_load_async_to_lds_b128 %1, %3, off"              \
                     :: "v"(LH), "v"(LL), "v"(gh_), "v"(gl_)                  \
                     : "memory");                                             \
    } while (0)

    v8f acc[2][4];
#pragma unroll
    for (int ms = 0; ms < 2; ++ms)
#pragma unroll
        for (int s = 0; s < 4; ++s) acc[ms][s] = (v8f)0.0f;

    // prologue: stage chunk 0 into buffer 0
    STAGE_ASYNC(ldsAh0, ldsAl0, 0);
    asm volatile("s_wait_asynccnt 0x0" ::: "memory");
    __syncthreads();

    for (int k0 = 0; k0 < DIN; k0 += KC) {
        const int cur = (k0 >> 5) & 1;

        // prefetch next chunk into the other buffer (overlaps with WMMAs)
        if (k0 + KC < DIN) {
            if (cur == 0) STAGE_ASYNC(ldsAh1, ldsAl1, k0 + KC);
            else          STAGE_ASYNC(ldsAh0, ldsAl0, k0 + KC);
        }

        // A fragments: K = 8h+[0..7] in VGPR0..3, K = 16+8h+[0..7] in VGPR4..7
        v16bf afrag[2][2];                  // [ms][hi=0 / lo=1]
#pragma unroll
        for (int ms = 0; ms < 2; ++ms) {
            const bf16_t* ra = &Ah[cur][ms * 16 + m][0];
            const bf16_t* rl = &Al[cur][ms * 16 + m][0];
            v8bf p0 = *(const v8bf*)(ra + 8 * h);
            v8bf p1 = *(const v8bf*)(ra + 16 + 8 * h);
            v8bf q0 = *(const v8bf*)(rl + 8 * h);
            v8bf q1 = *(const v8bf*)(rl + 16 + 8 * h);
#pragma unroll
            for (int j = 0; j < 8; ++j) {
                afrag[ms][0][j] = p0[j]; afrag[ms][0][8 + j] = p1[j];
                afrag[ms][1][j] = q0[j]; afrag[ms][1][8 + j] = q1[j];
            }
        }

#pragma unroll
        for (int s = 0; s < 4; ++s) {
            // B fragment: lane = column, K = 8h+[0..7], then 16+8h+[0..7]
            const size_t woff =
                (size_t)(ncol0 + (wave * 4 + s) * 16 + m) * DIN + k0 + 8 * h;
            v8bf b0 = *(const v8bf*)(Whb + woff);
            v8bf b1 = *(const v8bf*)(Whb + woff + 16);
            v8bf c0 = *(const v8bf*)(Wlb + woff);
            v8bf c1 = *(const v8bf*)(Wlb + woff + 16);
            v16bf bh, bl;
#pragma unroll
            for (int j = 0; j < 8; ++j) {
                bh[j] = b0[j]; bh[8 + j] = b1[j];
                bl[j] = c0[j]; bl[8 + j] = c1[j];
            }
#pragma unroll
            for (int ms = 0; ms < 2; ++ms) {
                acc[ms][s] = __builtin_amdgcn_wmma_f32_16x16x32_bf16(
                    false, afrag[ms][0], false, bh, (short)0, acc[ms][s], false, false);
                acc[ms][s] = __builtin_amdgcn_wmma_f32_16x16x32_bf16(
                    false, afrag[ms][1], false, bh, (short)0, acc[ms][s], false, false);
                acc[ms][s] = __builtin_amdgcn_wmma_f32_16x16x32_bf16(
                    false, afrag[ms][0], false, bl, (short)0, acc[ms][s], false, false);
            }
        }

        // next-chunk async writes landed; one barrier per chunk
        asm volatile("s_wait_asynccnt 0x0" ::: "memory");
        __syncthreads();
    }
#undef STAGE_ASYNC

    // C/D layout: lane l, VGPR j -> M = j + 8*(l>=16), N = l&15
#pragma unroll
    for (int ms = 0; ms < 2; ++ms)
#pragma unroll
        for (int s = 0; s < 4; ++s) {
            const int ncol = ncol0 + (wave * 4 + s) * 16 + m;
#pragma unroll
            for (int j = 0; j < 8; ++j) {
                const int row = slot0 + ms * 16 + h * 8 + j;
                Y[(size_t)row * DOUT + ncol] = acc[ms][s][j];
            }
        }
}

// ---------------- fallback: fused split (no workspace needed) ---------------
__global__ __launch_bounds__(BT)
void moe_gemm_fused(const float* __restrict__ X, const float* __restrict__ W,
                    const int* __restrict__ esz, const int* __restrict__ bidx,
                    float* __restrict__ Y, int E) {
    __shared__ float As[16][36];
    __shared__ int   tok[16];
    __shared__ int   s_expert;

    const int tid   = threadIdx.x;
    const int slot0 = blockIdx.x * 16;
    const int nblk  = blockIdx.y;

    if (tid < 16) tok[tid] = bidx[slot0 + tid];
    if (tid == 0) {
        int acc = 0, e = E - 1;
        for (int i = 0; i < E; ++i) { acc += esz[i]; if (slot0 < acc) { e = i; break; } }
        s_expert = e;
    }
    __syncthreads();

    const int lane = tid & 31, wave = tid >> 5;
    const int m = lane & 15, h = lane >> 4;
    const float* __restrict__ Wbase =
        W + ((size_t)s_expert * DOUT + (size_t)nblk * BN) * DIN;

    v8f acc[4];
#pragma unroll
    for (int s = 0; s < 4; ++s) acc[s] = (v8f)0.0f;

    for (int k0 = 0; k0 < DIN; k0 += KC) {
        __syncthreads();
        {
            const int r = tid >> 3, c = (tid & 7) * 4;
            const float4 v = *(const float4*)(X + (size_t)tok[r] * DIN + k0 + c);
            As[r][c + 0] = v.x; As[r][c + 1] = v.y;
            As[r][c + 2] = v.z; As[r][c + 3] = v.w;
        }
        __syncthreads();

        v16bf a_hi, a_lo;
#pragma unroll
        for (int v = 0; v < 8; ++v) {
            const int k = (v < 4) ? (8 * h + 2 * v) : (16 + 8 * h + 2 * (v - 4));
            const bf2 s0 = split1(As[m][k + 0]);
            const bf2 s1 = split1(As[m][k + 1]);
            a_hi[2 * v] = s0.hi; a_lo[2 * v] = s0.lo;
            a_hi[2 * v + 1] = s1.hi; a_lo[2 * v + 1] = s1.lo;
        }
#pragma unroll
        for (int s = 0; s < 4; ++s) {
            const int ncol = (wave * 4 + s) * 16 + m;
            const float* __restrict__ wrow = Wbase + (size_t)ncol * DIN + k0;
            v16bf b_hi, b_lo;
#pragma unroll
            for (int v = 0; v < 8; ++v) {
                const int k = (v < 4) ? (8 * h + 2 * v) : (16 + 8 * h + 2 * (v - 4));
                const bf2 s0 = split1(wrow[k + 0]);
                const bf2 s1 = split1(wrow[k + 1]);
                b_hi[2 * v] = s0.hi; b_lo[2 * v] = s0.lo;
                b_hi[2 * v + 1] = s1.hi; b_lo[2 * v + 1] = s1.lo;
            }
            acc[s] = __builtin_amdgcn_wmma_f32_16x16x32_bf16(
                false, a_hi, false, b_hi, (short)0, acc[s], false, false);
            acc[s] = __builtin_amdgcn_wmma_f32_16x16x32_bf16(
                false, a_lo, false, b_hi, (short)0, acc[s], false, false);
            acc[s] = __builtin_amdgcn_wmma_f32_16x16x32_bf16(
                false, a_hi, false, b_lo, (short)0, acc[s], false, false);
        }
    }
#pragma unroll
    for (int s = 0; s < 4; ++s) {
        const int ncol = nblk * BN + (wave * 4 + s) * 16 + m;
#pragma unroll
        for (int j = 0; j < 8; ++j)
            Y[(size_t)(slot0 + h * 8 + j) * DOUT + ncol] = acc[s][j];
    }
}

extern "C" void kernel_launch(void* const* d_in, const int* in_sizes, int n_in,
                              void* d_out, int out_size, void* d_ws, size_t ws_size,
                              hipStream_t stream) {
    const float* X    = (const float*)d_in[0];   // [T, 1024] f32
    const float* W    = (const float*)d_in[1];   // [E, 2048, 1024] f32
    const int*   esz  = (const int*)d_in[2];     // [E] i32
    const int*   bidx = (const int*)d_in[3];     // [N] i32
    float*       Y    = (float*)d_out;           // [N, 2048] f32

    const int    E  = in_sizes[2];
    const int    N  = in_sizes[3];               // 16384 routed slots
    const size_t nW = (size_t)in_sizes[1];
    const size_t nX = (size_t)in_sizes[0];
    const size_t need = (nW + nX) * 2 * sizeof(bf16_t);   // hi+lo for W and X

    if (ws_size >= need && (N % BM) == 0) {
        bf16_t* base = (bf16_t*)d_ws;
        bf16_t* Wh = base;
        bf16_t* Wl = base + nW;
        bf16_t* Xh = base + 2 * nW;
        bf16_t* Xl = base + 2 * nW + nX;

        const long nW4 = (long)(nW / 4), nX4 = (long)(nX / 4);
        split_f32_bf16<<<dim3((unsigned)((nW4 + 255) / 256)), 256, 0, stream>>>(W, Wh, Wl, nW4);
        split_f32_bf16<<<dim3((unsigned)((nX4 + 255) / 256)), 256, 0, stream>>>(X, Xh, Xl, nX4);

        dim3 grid(N / BM, DOUT / BN);            // (512, 8)
        moe_gemm_split<<<grid, BT, 0, stream>>>(Xh, Xl, Wh, Wl, esz, bidx, Y, E);
    } else {
        dim3 grid(N / 16, DOUT / BN);            // (1024, 8)
        moe_gemm_fused<<<grid, BT, 0, stream>>>(X, W, esz, bidx, Y, E);
    }
}
